// VQVAE_78941498900942
// MI455X (gfx1250) — compile-verified, matrix operand order checked by reading, not run on hardware
//
#include <hip/hip_runtime.h>
#include <hip/hip_bf16.h>

// ---------------------------------------------------------------------------
// VQ-VAE forward for MI455X (gfx1250, wave32, WMMA).
// Convs: implicit-GEMM on v_wmma_f32_16x16x32_bf16, bf16 NHWC activations,
// f32 accumulate, fused input-relu / bias / residual. Weight tiles staged with
// GLOBAL_LOAD_ASYNC_TO_LDS_B128 (ASYNCcnt) overlapping the im2col A-fill.
// VQ: WMMA score GEMM + cross-lane argmin. Losses: f32 atomics.
// ---------------------------------------------------------------------------

typedef __attribute__((ext_vector_type(16))) __bf16 v16bf;
typedef __attribute__((ext_vector_type(8)))  __bf16 v8bf;
typedef __attribute__((ext_vector_type(8)))  float  v8f;
typedef unsigned short u16;
typedef unsigned int   u32;

__device__ __forceinline__ float bf2f(u16 h) { return __uint_as_float(((u32)h) << 16); }
__device__ __forceinline__ u16 f2bf(float f) {
    u32 u = __float_as_uint(f);
    u32 r = u + 0x7FFFu + ((u >> 16) & 1u);   // round-to-nearest-even
    return (u16)(r >> 16);
}
// relu on two packed bf16 halves: zero any half whose sign bit is set
__device__ __forceinline__ u32 relu_pk(u32 u) {
    u32 m = ((u >> 15) & 0x00010001u) * 0xFFFFu;
    return u & ~m;
}
__device__ __forceinline__ v16bf cat8(v8bf lo, v8bf hi) {
    return __builtin_shufflevector(lo, hi, 0,1,2,3,4,5,6,7,8,9,10,11,12,13,14,15);
}

// LDS byte offset of a shared-memory pointer (addrspace(3) is 32-bit; the LDS
// aperture tag lives entirely in addr[63:32], so ptrtoint of the AS(3) pointer
// is the wave-relative LDS address the async-DMA VDST operand expects).
typedef __attribute__((address_space(3))) void lds_void_t;
__device__ __forceinline__ unsigned lds_addr_of(void* p) {
    return (unsigned)(unsigned long long)(lds_void_t*)p;
}
__device__ __forceinline__ void async_copy_b128(unsigned lds_addr, const void* g) {
    asm volatile("global_load_async_to_lds_b128 %0, %1, off"
                 :: "v"(lds_addr), "v"((unsigned long long)(size_t)g) : "memory");
}
__device__ __forceinline__ void wait_async0() {
    asm volatile("s_wait_asynccnt 0x0" ::: "memory");
}

// ---------------------------------------------------------------------------
// Implicit-GEMM convolution (stride 1, square, "same" size).
//   A: im2col of input, M = B*H*W rows, K = KH*KW*Cin (channel-fastest)
//   B: packed weights, [Cout][Kpad] rows
// 256 threads = 8 waves. Template tile shapes:
//   <8,1> : 128(M) x 16(N)  (Cout <= 16)
//   <4,1> :  64(M) x 32(N)  (Cout <= 32)
//   <4,2> :  64(M) x 64(N)  (2 accumulators / wave, A-fragment reuse)
// ---------------------------------------------------------------------------
#define AST 48  // LDS row stride in u16 (96B: 16B-aligned, banks spread)

template<int MW, int NW>
__global__ __launch_bounds__(256)
void conv_wmma_kernel(const u16* __restrict__ in, const u16* __restrict__ wk,
                      const float* __restrict__ bias, const u16* __restrict__ resid,
                      u16* __restrict__ out,
                      int Mtot, int H, int W, int Cin, int Cout,
                      int KH, int KW, int pad, int Kred, int Kpad, int relu_in)
{
    constexpr int MT   = MW * 16;         // rows per block: 64 or 128
    constexpr int NWAV = 8 / MW;          // waves along N: 2 or 1
    constexpr int BN   = NWAV * NW * 16;  // cols per block: 16 / 32 / 64
    constexpr int ACH  = MT / 64;         // A-tile 16B chunks per thread

    __shared__ u16 As[MT * AST];
    __shared__ u16 Bs[BN * AST];

    const int tid  = threadIdx.x;
    const int lane = tid & 31;
    const int wave = tid >> 5;
    const int mi   = wave / NWAV;
    const int ni   = wave % NWAV;
    const int m0   = blockIdx.x * MT;
    const int n0   = blockIdx.y * BN;
    const int HW   = H * W;

    v8f acc[NW];
#pragma unroll
    for (int t = 0; t < NW; ++t)
#pragma unroll
        for (int i = 0; i < 8; ++i) acc[t][i] = 0.0f;

    // per-chunk im2col row geometry (each thread owns ACH chunks of 8 K-elems)
    int r_row[ACH], r_kk[ACH], r_an[ACH], r_oh[ACH], r_ow[ACH];
#pragma unroll
    for (int cc = 0; cc < ACH; ++cc) {
        int chunk = tid + cc * 256;
        int row = chunk >> 2;
        r_row[cc] = row;
        r_kk[cc]  = (chunk & 3) * 8;
        int am  = m0 + row;
        int an  = am / HW;
        int ahw = am - an * HW;
        int aoh = ahw / W;
        r_an[cc] = an;
        r_oh[cc] = aoh;
        r_ow[cc] = ahw - aoh * W;
    }

    for (int k0 = 0; k0 < Kpad; k0 += 32) {
        __syncthreads();   // everyone done reading previous tiles

        // ---- B tile: async DMA global -> LDS (overlaps A im2col below) ----
        if (tid < BN * 4) {
            int row = tid >> 2;
            int kk  = (tid & 3) * 8;
            const u16* g = wk + (size_t)(n0 + row) * Kpad + k0 + kk;
            async_copy_b128(lds_addr_of(&Bs[row * AST + kk]), g);
        }

        // ---- A tile: im2col with fused relu ----
#pragma unroll
        for (int cc = 0; cc < ACH; ++cc) {
            int kbase = k0 + r_kk[cc];
            int am    = m0 + r_row[cc];
            uint4 v = {0u, 0u, 0u, 0u};
            if (((Cin & 7) == 0) && (kbase + 8 <= Kred)) {
                int khkw = kbase / Cin;
                int ci   = kbase - khkw * Cin;
                int kh   = khkw / KW;
                int kw   = khkw - kh * KW;
                int ih = r_oh[cc] - pad + kh, iw = r_ow[cc] - pad + kw;
                if (am < Mtot && (unsigned)ih < (unsigned)H && (unsigned)iw < (unsigned)W) {
                    const u16* p = in + (((size_t)r_an[cc] * H + ih) * W + iw) * (size_t)Cin + ci;
                    v = *reinterpret_cast<const uint4*>(p);
                }
                if (relu_in) { v.x = relu_pk(v.x); v.y = relu_pk(v.y); v.z = relu_pk(v.z); v.w = relu_pk(v.w); }
            } else {
                u16 tmp[8];
#pragma unroll
                for (int j = 0; j < 8; ++j) {
                    int k = kbase + j;
                    u16 val = 0;
                    if (am < Mtot && k < Kred) {
                        int khkw = k / Cin;
                        int ci   = k - khkw * Cin;
                        int kh   = khkw / KW;
                        int kw   = khkw - kh * KW;
                        int ih = r_oh[cc] - pad + kh, iw = r_ow[cc] - pad + kw;
                        if ((unsigned)ih < (unsigned)H && (unsigned)iw < (unsigned)W) {
                            val = in[(((size_t)r_an[cc] * H + ih) * W + iw) * (size_t)Cin + ci];
                            if (relu_in && (val & 0x8000u)) val = 0;
                        }
                    }
                    tmp[j] = val;
                }
                v.x = (u32)tmp[0] | ((u32)tmp[1] << 16);
                v.y = (u32)tmp[2] | ((u32)tmp[3] << 16);
                v.z = (u32)tmp[4] | ((u32)tmp[5] << 16);
                v.w = (u32)tmp[6] | ((u32)tmp[7] << 16);
            }
            *reinterpret_cast<uint4*>(&As[r_row[cc] * AST + r_kk[cc]]) = v;
        }

        wait_async0();     // B tile DMA complete (ASYNCcnt == 0)
        __syncthreads();

        // ---- WMMA: one A fragment reused across NW B tiles ----
        {
            const int zr = lane & 15;
            const int co = (lane >> 4) * 8;
            const u16* ap = &As[(mi * 16 + zr) * AST + co];
            v8bf alo = *reinterpret_cast<const v8bf*>(ap);
            v8bf ahi = *reinterpret_cast<const v8bf*>(ap + 16);
            v16bf a = cat8(alo, ahi);
#pragma unroll
            for (int t = 0; t < NW; ++t) {
                const u16* bp = &Bs[(ni * NW * 16 + t * 16 + zr) * AST + co];
                v8bf blo = *reinterpret_cast<const v8bf*>(bp);
                v8bf bhi = *reinterpret_cast<const v8bf*>(bp + 16);
                acc[t] = __builtin_amdgcn_wmma_f32_16x16x32_bf16(false, a, false, cat8(blo, bhi),
                                                                 (short)0, acc[t], false, false);
            }
        }
    }

    // ---- epilogue: bias (+ residual), store bf16 ----
    const int mbase = m0 + mi * 16 + ((lane >> 4) * 8);
#pragma unroll
    for (int t = 0; t < NW; ++t) {
        int col = n0 + ni * NW * 16 + t * 16 + (lane & 15);
        if (col < Cout) {
            float bv = bias[col];
#pragma unroll
            for (int r = 0; r < 8; ++r) {
                int m = mbase + r;
                if (m < Mtot) {
                    float v = acc[t][r] + bv;
                    size_t oi = (size_t)m * Cout + col;
                    if (resid) v += bf2f(resid[oi]);
                    out[oi] = f2bf(v);
                }
            }
        }
    }
}

// ---------------------------------------------------------------------------
// Vector quantizer: per block, 16 latent rows. argmax(z.e - 0.5|e|^2) over 512
// codes via WMMA, then gather q (fp32 codebook), write bf16 q, accumulate
// sum((q - relu(z))^2).
// ---------------------------------------------------------------------------
#define ZST 280  // LDS row stride in u16 (560B = 35*16, banks spread)

__global__ __launch_bounds__(256)
void vq_kernel(const u16* __restrict__ z, const u16* __restrict__ embb,
               const float* __restrict__ embf, const float* __restrict__ halfnorm,
               u16* __restrict__ q, float* __restrict__ accum)
{
    __shared__ u16   Zs[16 * ZST];
    __shared__ float redV[8][16];
    __shared__ int   redI[8][16];
    __shared__ int   rowIdx[16];
    __shared__ float sred[256];

    const int tid  = threadIdx.x;
    const int lane = tid & 31;
    const int wave = tid >> 5;
    const size_t rowBase = (size_t)blockIdx.x * 16;

    // stage 16 rows of z (relu fused)
    {
        int r = tid >> 4;
        int c = (tid & 15) * 16;
        const u16* src = z + (rowBase + r) * 256 + c;
        uint4 v0 = *reinterpret_cast<const uint4*>(src);
        uint4 v1 = *reinterpret_cast<const uint4*>(src + 8);
        v0.x = relu_pk(v0.x); v0.y = relu_pk(v0.y); v0.z = relu_pk(v0.z); v0.w = relu_pk(v0.w);
        v1.x = relu_pk(v1.x); v1.y = relu_pk(v1.y); v1.z = relu_pk(v1.z); v1.w = relu_pk(v1.w);
        *reinterpret_cast<uint4*>(&Zs[r * ZST + c])     = v0;
        *reinterpret_cast<uint4*>(&Zs[r * ZST + c + 8]) = v1;
    }
    __syncthreads();

    float best[8];
    int   bidx[8];
#pragma unroll
    for (int r = 0; r < 8; ++r) { best[r] = -3.0e38f; bidx[r] = 0; }

    const int zr = lane & 15;
    const int co = (lane >> 4) * 8;
#pragma unroll
    for (int t = 0; t < 4; ++t) {
        int nt = wave * 4 + t;   // column tile 0..31 (16 codes each)
        v8f acc;
#pragma unroll
        for (int i = 0; i < 8; ++i) acc[i] = 0.0f;
#pragma unroll
        for (int ks = 0; ks < 8; ++ks) {  // K = 256 = 8 * 32
            const u16* ap = &Zs[zr * ZST + ks * 32 + co];
            v8bf alo = *reinterpret_cast<const v8bf*>(ap);
            v8bf ahi = *reinterpret_cast<const v8bf*>(ap + 16);
            const u16* bp = embb + ((size_t)(nt * 16 + zr)) * 256 + ks * 32 + co;
            v8bf blo = *reinterpret_cast<const v8bf*>(bp);
            v8bf bhi = *reinterpret_cast<const v8bf*>(bp + 16);
            acc = __builtin_amdgcn_wmma_f32_16x16x32_bf16(false, cat8(alo, ahi),
                                                          false, cat8(blo, bhi),
                                                          (short)0, acc, false, false);
        }
        int colc = nt * 16 + zr;
        float hn = halfnorm[colc];
#pragma unroll
        for (int r = 0; r < 8; ++r) {
            float s = acc[r] - hn;
            if (s > best[r] || (s == best[r] && colc < bidx[r])) { best[r] = s; bidx[r] = colc; }
        }
    }
    // reduce across the 16 lanes that share rows (xor masks < 16 stay in group)
#pragma unroll
    for (int off = 8; off >= 1; off >>= 1) {
#pragma unroll
        for (int r = 0; r < 8; ++r) {
            float ov = __shfl_xor(best[r], off, 32);
            int   oi = __shfl_xor(bidx[r], off, 32);
            if (ov > best[r] || (ov == best[r] && oi < bidx[r])) { best[r] = ov; bidx[r] = oi; }
        }
    }
    if ((lane & 15) == 0) {
        int rb = (lane >> 4) * 8;
#pragma unroll
        for (int r = 0; r < 8; ++r) { redV[wave][rb + r] = best[r]; redI[wave][rb + r] = bidx[r]; }
    }
    __syncthreads();
    if (tid < 16) {
        float bv = redV[0][tid];
        int   bi = redI[0][tid];
        for (int w = 1; w < 8; ++w) {
            if (redV[w][tid] > bv || (redV[w][tid] == bv && redI[w][tid] < bi)) {
                bv = redV[w][tid]; bi = redI[w][tid];
            }
        }
        rowIdx[tid] = bi;
    }
    __syncthreads();

    // gather q (fp32 codebook), write bf16, accumulate sum((q - z)^2)
    float local = 0.0f;
    {
        int r  = tid >> 4;
        int c0 = (tid & 15) * 16;
        int e  = rowIdx[r];
        const float* ep = embf + (size_t)e * 256 + c0;
        u16* qp = q + (rowBase + r) * 256 + c0;
#pragma unroll
        for (int j = 0; j < 16; ++j) {
            float qv = ep[j];
            float zv = bf2f(Zs[r * ZST + c0 + j]);
            qp[j] = f2bf(qv);
            float d = qv - zv;
            local += d * d;
        }
    }
    sred[tid] = local;
    __syncthreads();
    for (int s = 128; s > 0; s >>= 1) {
        if (tid < s) sred[tid] += sred[tid + s];
        __syncthreads();
    }
    if (tid == 0) atomicAdd(accum, sred[0]);
}

// ---------------------------------------------------------------------------
// Small helper kernels
// ---------------------------------------------------------------------------
__global__ void prep_weight_kernel(const float* __restrict__ w, u16* __restrict__ out,
                                   int Cout, int Cin, int KH, int KW, int Kpad)
{
    int i = blockIdx.x * 256 + threadIdx.x;
    int total = Cout * Kpad;
    if (i >= total) return;
    int o = i / Kpad;
    int k = i - o * Kpad;
    int K = KH * KW * Cin;
    u16 v = 0;
    if (k < K) {
        int khkw = k / Cin;
        int ci   = k - khkw * Cin;
        int kh   = khkw / KW;
        int kw   = khkw - kh * KW;
        v = f2bf(w[(((size_t)o * Cin + ci) * KH + kh) * KW + kw]);
    }
    out[i] = v;
}

__global__ void convert_x_kernel(const float* __restrict__ x, u16* __restrict__ out, int total)
{
    int i = blockIdx.x * 256 + threadIdx.x;
    if (i >= total) return;
    // NCHW f32 -> NHWC bf16, H=W=256, C=3
    int c = i % 3;
    int p = i / 3;
    int hw = p & 65535;
    int n  = p >> 16;
    out[i] = f2bf(x[(((size_t)n * 3 + c) << 16) + hw]);
}

__global__ void prep_emb_kernel(const float* __restrict__ emb, u16* __restrict__ embb,
                                float* __restrict__ halfnorm)
{
    __shared__ float s[256];
    int e = blockIdx.x;
    int t = threadIdx.x;
    float v = emb[(size_t)e * 256 + t];
    embb[(size_t)e * 256 + t] = f2bf(v);
    s[t] = v * v;
    __syncthreads();
    for (int st = 128; st > 0; st >>= 1) {
        if (t < st) s[t] += s[t + st];
        __syncthreads();
    }
    if (t == 0) halfnorm[e] = 0.5f * s[0];
}

__global__ void maxpool_kernel(const u16* __restrict__ in, u16* __restrict__ out,
                               int H, int W, int C, int total)
{
    int i = blockIdx.x * 256 + threadIdx.x;
    if (i >= total) return;
    int c = i % C;
    int p = i / C;
    int Wo = W >> 1, Ho = H >> 1;
    int ow = p % Wo; p /= Wo;
    int oh = p % Ho;
    int n  = p / Ho;
    const u16* base = in + (((size_t)n * H + oh * 2) * W + ow * 2) * C + c;
    float a = bf2f(base[0]);
    float b = bf2f(base[C]);
    float d0 = bf2f(base[(size_t)W * C]);
    float d1 = bf2f(base[(size_t)W * C + C]);
    out[i] = f2bf(fmaxf(fmaxf(a, b), fmaxf(d0, d1)));
}

__global__ void upsample_kernel(const u16* __restrict__ in, u16* __restrict__ out,
                                int H, int W, int C, int total)
{
    int i = blockIdx.x * 256 + threadIdx.x;
    if (i >= total) return;
    int c = i % C;
    int p = i / C;
    int W2 = W * 2, H2 = H * 2;
    int ow = p % W2; p /= W2;
    int oh = p % H2;
    int n  = p / H2;
    out[i] = in[(((size_t)n * H + (oh >> 1)) * W + (ow >> 1)) * C + c];
}

__global__ void recon_loss_kernel(const u16* __restrict__ y, const float* __restrict__ x,
                                  float* __restrict__ accum, int M)
{
    __shared__ float s[256];
    int m = blockIdx.x * 256 + threadIdx.x;
    float local = 0.0f;
    if (m < M) {
        int hw = m & 65535;
        int n  = m >> 16;
#pragma unroll
        for (int c = 0; c < 3; ++c) {
            float yv = tanhf(bf2f(y[(size_t)m * 3 + c]));
            float xv = x[(((size_t)(n * 3 + c)) << 16) + hw];
            float d = yv - xv;
            local += d * d;
        }
    }
    s[threadIdx.x] = local;
    __syncthreads();
    for (int st = 128; st > 0; st >>= 1) {
        if (threadIdx.x < st) s[threadIdx.x] += s[threadIdx.x + st];
        __syncthreads();
    }
    if (threadIdx.x == 0) atomicAdd(accum, s[0]);
}

__global__ void finalize_kernel(const float* __restrict__ accum, float* __restrict__ out)
{
    // accum[0] = vq sum((q-z)^2), accum[1] = recon sum((tanh-x)^2)
    float vq    = 1.25f * accum[0] * (1.0f / (16384.0f * 256.0f));
    float recon = accum[1] * (1.0f / (16.0f * 3.0f * 256.0f * 256.0f));
    out[0] = vq + recon;
}

// ---------------------------------------------------------------------------
// Host orchestration
// ---------------------------------------------------------------------------
namespace {
struct LayerDef { int wi, bi, cin, cout, kh, kw, pad; };

// d_in flat index assumption: setup_inputs() dict INSERTION order,
// tuples (w,b) in order, None id-paths contribute no leaves:
//   0:x, 1,2:enc_in, 3..10:enc_b0.c, 11..18:enc_b1.c,
//   19,20:enc_b2.id 21..28:enc_b2.c, 29,30:enc_b3.id 31..38:enc_b3.c,
//   39:emb, 40,41:dec_b0.id 42..49:dec_b0.c, 50,51:dec_b1.id 52..59:dec_b1.c,
//   60..67:dec_b2.c, 68..75:dec_b3.c, 76,77:dec_out
static const LayerDef LD[38] = {
    {1, 2, 3, 64, 7, 7, 3},                                                     // 0 enc_in
    {3, 4, 64, 16, 3, 3, 1}, {5, 6, 16, 16, 3, 3, 1},                           // 1-4 enc_b0
    {7, 8, 16, 16, 3, 3, 1}, {9, 10, 16, 64, 1, 1, 0},
    {11, 12, 64, 16, 3, 3, 1}, {13, 14, 16, 16, 3, 3, 1},                       // 5-8 enc_b1
    {15, 16, 16, 16, 3, 3, 1}, {17, 18, 16, 64, 1, 1, 0},
    {19, 20, 64, 128, 1, 1, 0},                                                 // 9 enc_b2.id
    {21, 22, 64, 32, 3, 3, 1}, {23, 24, 32, 32, 3, 3, 1},                       // 10-13 enc_b2.c
    {25, 26, 32, 32, 3, 3, 1}, {27, 28, 32, 128, 1, 1, 0},
    {29, 30, 128, 256, 1, 1, 0},                                                // 14 enc_b3.id
    {31, 32, 128, 64, 3, 3, 1}, {33, 34, 64, 64, 3, 3, 1},                      // 15-18 enc_b3.c
    {35, 36, 64, 64, 3, 3, 1}, {37, 38, 64, 256, 1, 1, 0},
    {40, 41, 256, 128, 1, 1, 0},                                                // 19 dec_b0.id
    {42, 43, 256, 32, 1, 1, 0}, {44, 45, 32, 32, 3, 3, 1},                      // 20-23 dec_b0.c
    {46, 47, 32, 32, 3, 3, 1}, {48, 49, 32, 128, 3, 3, 1},
    {50, 51, 128, 64, 1, 1, 0},                                                 // 24 dec_b1.id
    {52, 53, 128, 16, 1, 1, 0}, {54, 55, 16, 16, 3, 3, 1},                      // 25-28 dec_b1.c
    {56, 57, 16, 16, 3, 3, 1}, {58, 59, 16, 64, 3, 3, 1},
    {60, 61, 64, 16, 1, 1, 0}, {62, 63, 16, 16, 3, 3, 1},                       // 29-32 dec_b2.c
    {64, 65, 16, 16, 3, 3, 1}, {66, 67, 16, 64, 3, 3, 1},
    {68, 69, 64, 16, 1, 1, 0}, {70, 71, 16, 16, 3, 3, 1},                       // 33-36 dec_b3.c
    {72, 73, 16, 16, 3, 3, 1}, {74, 75, 16, 64, 3, 3, 1},
    {76, 77, 64, 3, 1, 1, 0},                                                   // 37 dec_out
};
} // namespace

extern "C" void kernel_launch(void* const* d_in, const int* in_sizes, int n_in,
                              void* d_out, int out_size, void* d_ws, size_t ws_size,
                              hipStream_t stream)
{
    (void)in_sizes; (void)n_in; (void)out_size; (void)ws_size;

    size_t off = 0;
    auto alloc = [&](size_t bytes) -> void* {
        off = (off + 255) & ~(size_t)255;
        void* p = (char*)d_ws + off;
        off += bytes;
        return p;
    };

    const size_t M0 = (size_t)16 * 256 * 256;  // 1,048,576
    const size_t M1 = M0 / 4, M2 = M0 / 16, M3 = M0 / 64;  // M3 = 16384 latent rows

    // ---- weights: pack OIHW f32 -> [Cout][Kpad] bf16 (K channel-fastest) ----
    u16* wp[38];
    for (int i = 0; i < 38; ++i) {
        int K = LD[i].kh * LD[i].kw * LD[i].cin;
        int Kpad = (K + 31) & ~31;
        size_t total = (size_t)LD[i].cout * Kpad;
        wp[i] = (u16*)alloc(total * 2);
        prep_weight_kernel<<<(unsigned)((total + 255) / 256), 256, 0, stream>>>(
            (const float*)d_in[LD[i].wi], wp[i], LD[i].cout, LD[i].cin, LD[i].kh, LD[i].kw, Kpad);
    }

    // ---- activation buffers (bf16 NHWC) ----
    u16* bufA  = (u16*)alloc(M0 * 64 * 2);
    u16* bufB  = (u16*)alloc(M0 * 64 * 2);
    u16* bufC  = (u16*)alloc(M0 * 16 * 2);
    u16* bufD  = (u16*)alloc(M0 * 16 * 2);
    u16* bufE  = (u16*)alloc(M0 * 16 * 2);
    u16* xb    = (u16*)alloc(M0 * 3 * 2);
    u16* yout  = (u16*)alloc(M0 * 3 * 2);
    u16* qbuf  = (u16*)alloc(M3 * 256 * 2);
    u16* embb  = (u16*)alloc(512 * 256 * 2);
    float* hnorm = (float*)alloc(512 * 4);
    float* accum = (float*)alloc(2 * 4);

    hipMemsetAsync(accum, 0, 2 * sizeof(float), stream);

    prep_emb_kernel<<<512, 256, 0, stream>>>((const float*)d_in[39], embb, hnorm);
    convert_x_kernel<<<(unsigned)((M0 * 3 + 255) / 256), 256, 0, stream>>>(
        (const float*)d_in[0], xb, (int)(M0 * 3));

    auto runConv = [&](int li, const u16* in, u16* out, const u16* resid,
                       int Mtot, int Hc, int relu_in) {
        const LayerDef& L = LD[li];
        int K = L.kh * L.kw * L.cin;
        int Kpad = (K + 31) & ~31;
        const float* bias = (const float*)d_in[L.bi];
        if (L.cout <= 16) {
            dim3 g(Mtot / 128, (L.cout + 15) / 16);
            conv_wmma_kernel<8, 1><<<g, 256, 0, stream>>>(in, wp[li], bias, resid, out,
                Mtot, Hc, Hc, L.cin, L.cout, L.kh, L.kw, L.pad, K, Kpad, relu_in);
        } else if (L.cout <= 32) {
            dim3 g(Mtot / 64, (L.cout + 31) / 32);
            conv_wmma_kernel<4, 1><<<g, 256, 0, stream>>>(in, wp[li], bias, resid, out,
                Mtot, Hc, Hc, L.cin, L.cout, L.kh, L.kw, L.pad, K, Kpad, relu_in);
        } else {
            dim3 g(Mtot / 64, (L.cout + 63) / 64);
            conv_wmma_kernel<4, 2><<<g, 256, 0, stream>>>(in, wp[li], bias, resid, out,
                Mtot, Hc, Hc, L.cin, L.cout, L.kh, L.kw, L.pad, K, Kpad, relu_in);
        }
    };
    // enc block: relu-conv3x3, relu-conv3x3, relu-conv3x3, relu-conv1x1 (+id)
    // dec block: relu-conv1x1, relu-conv3x3, relu-conv3x3, relu-conv3x3 (+id)
    auto resBlock = [&](int li0, int lid, u16* x, u16* outb, u16* t1, u16* t2,
                        u16* idb, int Mtot, int Hc) {
        const u16* resid = x;
        if (lid >= 0) { runConv(lid, x, idb, nullptr, Mtot, Hc, 0); resid = idb; }
        runConv(li0 + 0, x,  t1, nullptr, Mtot, Hc, 1);
        runConv(li0 + 1, t1, t2, nullptr, Mtot, Hc, 1);
        runConv(li0 + 2, t2, t1, nullptr, Mtot, Hc, 1);
        runConv(li0 + 3, t1, outb, resid, Mtot, Hc, 1);
    };
    auto pool = [&](const u16* in, u16* out, int Hc, int C) {
        int total = (int)((size_t)16 * (Hc / 2) * (Hc / 2) * C);
        maxpool_kernel<<<(total + 255) / 256, 256, 0, stream>>>(in, out, Hc, Hc, C, total);
    };
    auto up2 = [&](const u16* in, u16* out, int Hc, int C) {
        int total = (int)((size_t)16 * Hc * 2 * Hc * 2 * C);
        upsample_kernel<<<(total + 255) / 256, 256, 0, stream>>>(in, out, Hc, Hc, C, total);
    };

    // ---------------- encoder ----------------
    runConv(0, xb, bufA, nullptr, (int)M0, 256, 0);                 // enc_in 7x7
    resBlock(1, -1, bufA, bufB, bufC, bufD, nullptr, (int)M0, 256); // enc_b0
    pool(bufB, bufA, 256, 64);
    resBlock(5, -1, bufA, bufB, bufC, bufD, nullptr, (int)M1, 128); // enc_b1
    pool(bufB, bufA, 128, 64);
    resBlock(10, 9, bufA, bufB, bufC, bufD, bufE, (int)M2, 64);     // enc_b2
    pool(bufB, bufA, 64, 128);
    resBlock(15, 14, bufA, bufB, bufC, bufD, bufE, (int)M3, 32);    // enc_b3

    // ---------------- vector quantizer (relu fused on z) ----------------
    vq_kernel<<<(unsigned)(M3 / 16), 256, 0, stream>>>(bufB, embb,
        (const float*)d_in[39], hnorm, qbuf, accum);

    // ---------------- decoder ----------------
    resBlock(20, 19, qbuf, bufA, bufC, bufD, bufE, (int)M3, 32);    // dec_b0
    up2(bufA, bufB, 32, 128);
    resBlock(25, 24, bufB, bufA, bufC, bufD, bufE, (int)M2, 64);    // dec_b1
    up2(bufA, bufB, 64, 64);
    resBlock(29, -1, bufB, bufA, bufC, bufD, nullptr, (int)M1, 128);// dec_b2
    up2(bufA, bufB, 128, 64);
    resBlock(33, -1, bufB, bufA, bufC, bufD, nullptr, (int)M0, 256);// dec_b3
    runConv(37, bufA, yout, nullptr, (int)M0, 256, 1);              // relu + dec_out 1x1

    // ---------------- losses ----------------
    recon_loss_kernel<<<(unsigned)((M0 + 255) / 256), 256, 0, stream>>>(
        yout, (const float*)d_in[0], accum + 1, (int)M0);
    finalize_kernel<<<1, 1, 0, stream>>>(accum, (float*)d_out);
}
